// DeformConv2d_53214644798137
// MI455X (gfx1250) — matrix operand order repaired and checked by reading, not compiled, not run on hardware
//
#include <hip/hip_runtime.h>
#include <math.h>

// ---------------------------------------------------------------------------
// Deformable conv v2 (DCN) for MI455X / gfx1250, wave32 + WMMA f16.
//   K0: weight [O,C,3,3] f32 -> Ah f16, layout Ah[o][k*256+c]
//   K1: fused heads (dw7x7 + pw1x1 for offset & mask) -> bilinear corner table
//   K2: fused deform-im2col (double-buffered LDS tile) + WMMA GEMM
//       M=256, N=25088, K=2304; block tile M=256 x N=32, 16 waves,
//       2 WMMAs/wave/chunk sharing one A fragment, 1 barrier per chunk.
// ---------------------------------------------------------------------------

typedef _Float16 v2h  __attribute__((ext_vector_type(2)));
typedef _Float16 v8h  __attribute__((ext_vector_type(8)));
typedef _Float16 v16h __attribute__((ext_vector_type(16)));
typedef float    v8f  __attribute__((ext_vector_type(8)));

constexpr int B_  = 8;
constexpr int C_  = 256;
constexpr int H_  = 56;
constexpr int W_  = 56;
constexpr int O_  = 256;
constexpr int K_  = 9;
constexpr int HW_ = H_ * W_;          // 3136
constexpr int CK_ = C_ * K_;          // 2304
constexpr int N_  = B_ * HW_;         // 25088 (multiple of 32)
constexpr int NCHUNK_ = CK_ / 32;     // 72 K-chunks

constexpr size_t AH_BYTES  = (size_t)O_ * CK_ * 2;        // 1,179,648
constexpr size_t CI_BYTES  = (size_t)B_ * K_ * HW_ * 16;  // 3,612,672
// total ws need = AH_BYTES + 2*CI_BYTES ~= 8.4 MB

// ---------------------------------------------------------------------------
// K0: convert weight[(o*C+c)*9+k] f32 -> Ah[o*2304 + k*256 + c] f16
// ---------------------------------------------------------------------------
__global__ __launch_bounds__(256)
void wconv_kernel(const float* __restrict__ w, _Float16* __restrict__ Ah) {
    int idx = blockIdx.x * 256 + threadIdx.x;
    if (idx >= O_ * CK_) return;
    int o = idx / CK_;
    int r = idx - o * CK_;
    int k = r >> 8;        // r / 256
    int c = r & 255;
    Ah[idx] = (_Float16)w[(o * C_ + c) * K_ + k];
}

// ---------------------------------------------------------------------------
// K1: fused offset/mask heads + bilinear corner table.
// One block (256 threads) per output position (b,ho,wo).
// corner_i[e] = 4 clamped in-plane offsets, corner_w[e] = 4 weights * mask
// with per-corner validity folded in; e = (b*9+k)*3136 + hw.
// ---------------------------------------------------------------------------
__global__ __launch_bounds__(256)
void heads_kernel(const float* __restrict__ x,
                  const float* __restrict__ dw_off_w, const float* __restrict__ dw_off_b,
                  const float* __restrict__ pw_off_w, const float* __restrict__ pw_off_b,
                  const float* __restrict__ dw_msk_w, const float* __restrict__ dw_msk_b,
                  const float* __restrict__ pw_msk_w, const float* __restrict__ pw_msk_b,
                  int4*  __restrict__ corner_i,
                  float4* __restrict__ corner_w)
{
    __shared__ float s_off[C_];
    __shared__ float s_msk[C_];
    __shared__ float s_head[2 * K_ + K_];   // 18 offsets + 9 masks

    const int pos = blockIdx.x;
    const int b   = pos / HW_;
    const int hw  = pos - b * HW_;
    const int ho  = hw / W_;
    const int wop = hw - ho * W_;
    const int c   = threadIdx.x;

    // depthwise 7x7, pad 3, for both heads (share the x loads)
    float acc_o = dw_off_b[c];
    float acc_m = dw_msk_b[c];
    const float* xb  = x + (size_t)(b * C_ + c) * HW_;
    const float* w7o = dw_off_w + c * 49;
    const float* w7m = dw_msk_w + c * 49;
    #pragma unroll
    for (int dy = -3; dy <= 3; ++dy) {
        int yy = ho + dy;
        if (yy < 0 || yy >= H_) continue;
        #pragma unroll
        for (int dx = -3; dx <= 3; ++dx) {
            int xx = wop + dx;
            if (xx < 0 || xx >= W_) continue;
            float xv = xb[yy * W_ + xx];
            int   wi = (dy + 3) * 7 + (dx + 3);
            acc_o = fmaf(xv, w7o[wi], acc_o);
            acc_m = fmaf(xv, w7m[wi], acc_m);
        }
    }
    s_off[c] = acc_o;
    s_msk[c] = acc_m;
    __syncthreads();

    // pointwise 1x1
    const int t = threadIdx.x;
    if (t < 18) {
        float s = pw_off_b[t];
        const float* pw = pw_off_w + t * C_;
        for (int cc = 0; cc < C_; ++cc) s = fmaf(s_off[cc], pw[cc], s);
        s_head[t] = s;
    } else if (t >= 32 && t < 32 + K_) {
        int j = t - 32;
        float s = pw_msk_b[j];
        const float* pw = pw_msk_w + j * C_;
        for (int cc = 0; cc < C_; ++cc) s = fmaf(s_msk[cc], pw[cc], s);
        s_head[18 + j] = s;
    }
    __syncthreads();

    // bilinear corner table per tap
    if (t < K_) {
        int kh = t / 3, kw = t - kh * 3;
        float ys = (float)(ho - 1 + kh) + s_head[2 * t + 0];   // dy channel
        float xs = (float)(wop - 1 + kw) + s_head[2 * t + 1];  // dx channel
        float m  = s_head[18 + t];
        float y0f = floorf(ys), x0f = floorf(xs);
        float wy = ys - y0f, wx = xs - x0f;
        int iy0 = (int)y0f, ix0 = (int)x0f;
        int iy1 = iy0 + 1,  ix1 = ix0 + 1;

        float w00 = (1.f - wy) * (1.f - wx);
        float w01 = (1.f - wy) * wx;
        float w10 = wy * (1.f - wx);
        float w11 = wy * wx;

        auto vld = [](int yv, int xv) -> float {
            return (yv >= 0 && yv < H_ && xv >= 0 && xv < W_) ? 1.f : 0.f;
        };
        auto cl = [](int v, int hi) { return v < 0 ? 0 : (v > hi ? hi : v); };

        int yc0 = cl(iy0, H_ - 1), yc1 = cl(iy1, H_ - 1);
        int xc0 = cl(ix0, W_ - 1), xc1 = cl(ix1, W_ - 1);

        size_t e = (size_t)(b * K_ + t) * HW_ + hw;
        corner_i[e] = make_int4(yc0 * W_ + xc0, yc0 * W_ + xc1,
                                yc1 * W_ + xc0, yc1 * W_ + xc1);
        corner_w[e] = make_float4(w00 * m * vld(iy0, ix0), w01 * m * vld(iy0, ix1),
                                  w10 * m * vld(iy1, ix0), w11 * m * vld(iy1, ix1));
    }
}

// ---------------------------------------------------------------------------
// K2: fused deform-sample + GEMM via v_wmma_f32_16x16x32_f16.
// Block = 512 threads = 16 waves; block tile M=256 x N=32.
// Double-buffered LDS B tile (32 ck x 32 n, f16): build chunk i+1 into
// buffer (i+1)&1 while WMMA-ing chunk i from buffer i&1 -> one barrier/chunk.
// Each chunk has a single deform tap k (CK layout ck = k*256 + c, 256%32==0),
// so bilinear corner data is loaded once per (n,k) and reused for 32 channels.
// ---------------------------------------------------------------------------
__global__ __launch_bounds__(512)
void dcn_gemm_kernel(const float* __restrict__ x,
                     const _Float16* __restrict__ Ah,
                     const int4*   __restrict__ corner_i,
                     const float4* __restrict__ corner_w,
                     const float*  __restrict__ bias,
                     float* __restrict__ out)
{
    // N-major tiles: Bt[buf][n (0..31)][ck_local]; row stride 40 halves (80 B)
    // keeps 16 B alignment and spreads LDS banks. 2 x 32 x 40 x 2B = 5 KB.
    __shared__ __align__(16) _Float16 Bt[2][32][40];

    const int n0   = blockIdx.x * 32;
    const int t    = threadIdx.x;
    const int lane = t & 31;
    const int wave = t >> 5;             // 0..15
    const int o_base = wave * 16;

    // ---- build-role indices: thread -> (n32, 2 adjacent channels) ----
    const int bn   = t & 31;             // n within 32-wide tile
    const int c2   = (t >> 5) * 2;       // 0,2,..,30 channel pair base
    const int n_g  = n0 + bn;
    const int bb   = n_g / HW_;
    const int bhw  = n_g - bb * HW_;

    // ---- mma-role indices (ISA 7.12.2 layouts) ----
    const int mm = lane & 15;            // A row / D column index
    const int hi = lane >> 4;            // half-wave select

    // accumulators (two N sub-tiles), init with bias:
    // D VGPR r <-> M = r + 8*hi, N = lane&15
    v8f acc0, acc1;
    #pragma unroll
    for (int r = 0; r < 8; ++r) {
        float bv = bias[o_base + hi * 8 + r];
        acc0[r] = bv;
        acc1[r] = bv;
    }

    const _Float16* arow = Ah + (size_t)(o_base + mm) * CK_;

    // corner-data cache for the build role (k changes every 8 chunks)
    int   cur_k = -1;
    int4  ci;
    float4 cw;

    auto build = [&](int chunk) {
        int k = chunk >> 3;              // tap
        int cc = chunk & 7;              // channel chunk within tap
        if (k != cur_k) {
            cur_k = k;
            size_t e = (size_t)(bb * K_ + k) * HW_ + bhw;
            ci = corner_i[e];
            cw = corner_w[e];
        }
        int c0 = cc * 32 + c2;
        const float* xp0 = x + (size_t)(bb * C_ + c0) * HW_;
        const float* xp1 = xp0 + HW_;
        float v0 = cw.x * xp0[ci.x] + cw.y * xp0[ci.y]
                 + cw.z * xp0[ci.z] + cw.w * xp0[ci.w];
        float v1 = cw.x * xp1[ci.x] + cw.y * xp1[ci.y]
                 + cw.z * xp1[ci.z] + cw.w * xp1[ci.w];
        v2h p;
        p[0] = (_Float16)v0;
        p[1] = (_Float16)v1;
        *(v2h*)&Bt[chunk & 1][bn][c2] = p;   // one 4 B ds_store
    };

    auto mma = [&](int chunk) {
        int k  = chunk >> 3;
        int cc = chunk & 7;
        int ck0 = k * C_ + cc * 32;
        // A fragment: lane holds row (o_base+mm),
        // K = hi*8..hi*8+7 and 16+hi*8..16+hi*8+7 (two 16 B loads)
        v8h a_lo = *(const v8h*)(arow + ck0 + hi * 8);
        v8h a_hi = *(const v8h*)(arow + ck0 + 16 + hi * 8);
        v16h af = __builtin_shufflevector(a_lo, a_hi,
                  0,1,2,3,4,5,6,7,8,9,10,11,12,13,14,15);
        // speculative prefetch of the next A chunk (global_prefetch_b8)
        __builtin_prefetch((const void*)(arow + ck0 + 32), 0, 3);

        // B fragments: lane holds N = mm, K = hi*8..hi*8+7 and 16+hi*8..16+hi*8+7
        const v8h* bp0 = (const v8h*)&Bt[chunk & 1][mm][hi * 8];
        v16h bf0 = __builtin_shufflevector(bp0[0], bp0[2],
                   0,1,2,3,4,5,6,7,8,9,10,11,12,13,14,15);
        acc0 = __builtin_amdgcn_wmma_f32_16x16x32_f16(
                   false, af, false, bf0, (short)0, acc0, false, false);

        const v8h* bp1 = (const v8h*)&Bt[chunk & 1][16 + mm][hi * 8];
        v16h bf1 = __builtin_shufflevector(bp1[0], bp1[2],
                   0,1,2,3,4,5,6,7,8,9,10,11,12,13,14,15);
        acc1 = __builtin_amdgcn_wmma_f32_16x16x32_f16(
                   false, af, false, bf1, (short)0, acc1, false, false);
    };

    // software pipeline: build(i+1) overlaps mma(i); single barrier per chunk
    build(0);
    __syncthreads();
    for (int chunk = 0; chunk < NCHUNK_; ++chunk) {
        if (chunk + 1 < NCHUNK_) build(chunk + 1);
        mma(chunk);
        __syncthreads();
    }

    // store D: two N sub-tiles per wave
    #pragma unroll
    for (int sub = 0; sub < 2; ++sub) {
        const int sn  = n0 + sub * 16 + mm;
        const int sb  = sn / HW_;
        const int shw = sn - sb * HW_;
        const v8f& a = sub ? acc1 : acc0;
        #pragma unroll
        for (int r = 0; r < 8; ++r) {
            out[((size_t)(sb * O_ + o_base + hi * 8 + r)) * HW_ + shw] = a[r];
        }
    }
}

// ---------------------------------------------------------------------------
extern "C" void kernel_launch(void* const* d_in, const int* in_sizes, int n_in,
                              void* d_out, int out_size, void* d_ws, size_t ws_size,
                              hipStream_t stream) {
    const float* x        = (const float*)d_in[0];
    const float* dw_off_w = (const float*)d_in[1];
    const float* dw_off_b = (const float*)d_in[2];
    const float* pw_off_w = (const float*)d_in[3];
    const float* pw_off_b = (const float*)d_in[4];
    const float* dw_msk_w = (const float*)d_in[5];
    const float* dw_msk_b = (const float*)d_in[6];
    const float* pw_msk_w = (const float*)d_in[7];
    const float* pw_msk_b = (const float*)d_in[8];
    const float* weight   = (const float*)d_in[9];
    const float* bias     = (const float*)d_in[10];
    float* out = (float*)d_out;

    char* ws = (char*)d_ws;
    _Float16* Ah  = (_Float16*)ws;                            // 1.125 MB
    int4*     ci  = (int4*)  (ws + AH_BYTES);                 // 3.45 MB
    float4*   cwt = (float4*)(ws + AH_BYTES + CI_BYTES);      // 3.45 MB

    (void)in_sizes; (void)n_in; (void)out_size; (void)ws_size;

    wconv_kernel<<<(O_ * CK_ + 255) / 256, 256, 0, stream>>>(weight, Ah);
    heads_kernel<<<B_ * HW_, 256, 0, stream>>>(x,
        dw_off_w, dw_off_b, pw_off_w, pw_off_b,
        dw_msk_w, dw_msk_b, pw_msk_w, pw_msk_b, ci, cwt);
    dcn_gemm_kernel<<<N_ / 32, 512, 0, stream>>>(x, Ah, ci, cwt, bias, out);
}
